// GATStack_19679540150894
// MI455X (gfx1250) — compile-verified, matrix operand order checked by reading, not run on hardware
//
#include <hip/hip_runtime.h>
#include <hip/hip_bf16.h>
#include <math.h>

typedef __attribute__((ext_vector_type(16))) _Float16 v16h;
typedef __attribute__((ext_vector_type(8)))  float    v8f;

#define NNODES 50000
#define NEDGES 1600000
#define DH 128
#define DOUT 64
#define NEG_SLOPE 0.2f

// ---------------------------------------------------------------------------
// WMMA GEMM: Y[n, j] = sum_k X[n, k] * W[j, k] + bias[j]
// X: [n_rows, 128] f32, W: [out_dim, 128] f32 (row-major), Y: [n_rows, out_dim]
// One block = 16 rows of X staged to LDS as f16. Wave w handles output cols
// [w*16, w*16+16). blockDim.x = 32 * (out_dim/16). grid.x = n_rows/16.
// ---------------------------------------------------------------------------
__global__ void gemm_wmma_f16(const float* __restrict__ X,
                              const float* __restrict__ W,
                              const float* __restrict__ bias,
                              float* __restrict__ Y,
                              int out_dim) {
    __shared__ _Float16 shx[16 * DH];
    const int row_base = blockIdx.x * 16;

    // Cooperative stage: 16 rows x 128 cols of X -> f16 LDS
    for (int idx = threadIdx.x; idx < 16 * DH; idx += blockDim.x) {
        int r = idx >> 7;
        int c = idx & 127;
        shx[idx] = (_Float16)X[(size_t)(row_base + r) * DH + c];
    }
    __syncthreads();

    const int wv   = threadIdx.x >> 5;   // N-tile index
    const int lane = threadIdx.x & 31;
    const int hi   = lane >> 4;          // lane group (K split)
    const int m    = lane & 15;          // A row within tile / D col within tile
    const int j    = wv * 16 + m;        // global output column

    v8f c = {};
#pragma unroll
    for (int kb = 0; kb < 4; ++kb) {     // K = 128 in chunks of 32
        v16h a, b;
#pragma unroll
        for (int i = 0; i < 16; ++i) {
            // 16-bit A/B fragment layout (ISA 7.12.2): element i of lane maps to
            // K = kb*32 + i + (i>=8 ? 8 : 0) + (lane>=16 ? 8 : 0)
            int k = kb * 32 + i + ((i >= 8) ? 8 : 0) + hi * 8;
            a[i] = shx[m * DH + k];
            b[i] = (_Float16)W[(size_t)j * DH + k];
        }
        c = __builtin_amdgcn_wmma_f32_16x16x32_f16(
                /*neg_a=*/false, a, /*neg_b=*/false, b,
                /*c_mod=*/(short)0, c, /*reuse_a=*/false, /*reuse_b=*/false);
    }

    const float bj = bias[j];
#pragma unroll
    for (int r = 0; r < 8; ++r) {
        int row = r + hi * 8;            // D layout: VGPR r -> M=r (+8 for hi lanes)
        Y[(size_t)(row_base + row) * out_dim + j] = c[r] + bj;
    }
}

// ---------------------------------------------------------------------------
// Per-layer init: acc = 0, maxv = -inf, denom = 0
// ---------------------------------------------------------------------------
__global__ void layer_init(float* __restrict__ acc,
                           float* __restrict__ maxv,
                           float* __restrict__ denom) {
    int tid = blockIdx.x * blockDim.x + threadIdx.x;
    if (tid < NNODES * DH) acc[tid] = 0.0f;
    if (tid < NNODES) {
        maxv[tid]  = __int_as_float(0xff800000u);  // -inf
        denom[tid] = 0.0f;
    }
}

// Monotonic-int trick for float atomic max (maxv pre-initialized to -inf).
__device__ inline void atomicMaxFloat(float* addr, float val) {
    int iv = __float_as_int(val);
    if (iv >= 0) atomicMax((int*)addr, iv);
    else         atomicMin((unsigned int*)addr, (unsigned int)iv);
}

// ---------------------------------------------------------------------------
// E1: one wave per edge. logit = att . leaky_relu(xl[src] + xr[dst]);
// segment max into maxv[dst].
// ---------------------------------------------------------------------------
__global__ void edge_logits(const float* __restrict__ xl,
                            const float* __restrict__ xr,
                            const float* __restrict__ att,
                            const int* __restrict__ src,
                            const int* __restrict__ dst,
                            float* __restrict__ logits,
                            float* __restrict__ maxv) {
    int wave = (blockIdx.x * blockDim.x + threadIdx.x) >> 5;
    int lane = threadIdx.x & 31;
    if (wave >= NEDGES) return;
    int s = src[wave], d = dst[wave];
    float4 a = ((const float4*)(xl + (size_t)s * DH))[lane];
    float4 b = ((const float4*)(xr + (size_t)d * DH))[lane];
    float4 w = ((const float4*)att)[lane];
    float acc = 0.0f, v;
    v = a.x + b.x; acc += w.x * (v > 0.0f ? v : NEG_SLOPE * v);
    v = a.y + b.y; acc += w.y * (v > 0.0f ? v : NEG_SLOPE * v);
    v = a.z + b.z; acc += w.z * (v > 0.0f ? v : NEG_SLOPE * v);
    v = a.w + b.w; acc += w.w * (v > 0.0f ? v : NEG_SLOPE * v);
#pragma unroll
    for (int off = 16; off > 0; off >>= 1)
        acc += __shfl_xor(acc, off, 32);
    if (lane == 0) {
        logits[wave] = acc;
        atomicMaxFloat(&maxv[d], acc);
    }
}

// ---------------------------------------------------------------------------
// E2: one thread per edge. ex = exp(logit - max[dst]); denom[dst] += ex.
// ---------------------------------------------------------------------------
__global__ void edge_exp(const float* __restrict__ logits,
                         const int* __restrict__ dst,
                         const float* __restrict__ maxv,
                         float* __restrict__ exv,
                         float* __restrict__ denom) {
    int e = blockIdx.x * blockDim.x + threadIdx.x;
    if (e >= NEDGES) return;
    int d = dst[e];
    float ex = expf(logits[e] - maxv[d]);
    exv[e] = ex;
    atomicAdd(&denom[d], ex);
}

// ---------------------------------------------------------------------------
// E3: one wave per edge. alpha = ex/denom[dst]; acc[dst] += alpha * xl[src].
// ---------------------------------------------------------------------------
__global__ void edge_scatter(const float* __restrict__ xl,
                             const int* __restrict__ src,
                             const int* __restrict__ dst,
                             const float* __restrict__ exv,
                             const float* __restrict__ denom,
                             float* __restrict__ acc) {
    int wave = (blockIdx.x * blockDim.x + threadIdx.x) >> 5;
    int lane = threadIdx.x & 31;
    if (wave >= NEDGES) return;
    int s = src[wave], d = dst[wave];
    float alpha = exv[wave] / denom[d];
    float4 a = ((const float4*)(xl + (size_t)s * DH))[lane];
    float* p = acc + (size_t)d * DH + 4 * lane;
    atomicAdd(p + 0, a.x * alpha);
    atomicAdd(p + 1, a.y * alpha);
    atomicAdd(p + 2, a.z * alpha);
    atomicAdd(p + 3, a.w * alpha);
}

// ---------------------------------------------------------------------------
// N1: xcur = relu(acc + layer_bias)
// ---------------------------------------------------------------------------
__global__ void bias_relu(const float* __restrict__ acc,
                          const float* __restrict__ bias,
                          float* __restrict__ xcur) {
    int tid = blockIdx.x * blockDim.x + threadIdx.x;
    if (tid >= NNODES * DH) return;
    float v = acc[tid] + bias[tid & 127];
    xcur[tid] = v > 0.0f ? v : 0.0f;
}

extern "C" void kernel_launch(void* const* d_in, const int* in_sizes, int n_in,
                              void* d_out, int out_size, void* d_ws, size_t ws_size,
                              hipStream_t stream) {
    const float* x    = (const float*)d_in[0];
    const int*   eidx = (const int*)  d_in[1];   // [2, E]
    const float* Wl   = (const float*)d_in[2];   // [3,128,128]
    const float* bl   = (const float*)d_in[3];   // [3,128]
    const float* Wr   = (const float*)d_in[4];
    const float* br   = (const float*)d_in[5];
    const float* att  = (const float*)d_in[6];   // [3,128]
    const float* bias = (const float*)d_in[7];   // [3,128]
    const float* W1   = (const float*)d_in[8];   // [128,128]
    const float* b1   = (const float*)d_in[9];
    const float* W2   = (const float*)d_in[10];  // [64,128]
    const float* b2   = (const float*)d_in[11];
    float* out = (float*)d_out;

    const int* src = eidx;
    const int* dst = eidx + NEDGES;

    // Workspace layout (floats)
    float* ws    = (float*)d_ws;
    float* xcur  = ws;                        // N*128
    float* xl    = xcur  + (size_t)NNODES * DH;
    float* xr    = xl    + (size_t)NNODES * DH;
    float* acc   = xr    + (size_t)NNODES * DH;
    float* logit = acc   + (size_t)NNODES * DH;
    float* exv   = logit + (size_t)NEDGES;
    float* maxv  = exv   + (size_t)NEDGES;
    float* denom = maxv  + (size_t)NNODES;

    (void)in_sizes; (void)n_in; (void)out_size; (void)ws_size;

    hipMemcpyAsync(xcur, x, (size_t)NNODES * DH * sizeof(float),
                   hipMemcpyDeviceToDevice, stream);

    const int gemm_blocks = NNODES / 16;            // 3125
    const int edge_wave_grid = (NEDGES * 32 + 255) / 256;
    const int edge_thr_grid  = (NEDGES + 255) / 256;
    const int node_grid      = (NNODES * DH + 255) / 256;

    for (int l = 0; l < 3; ++l) {
        const float* Wl_l = Wl + (size_t)l * DH * DH;
        const float* Wr_l = Wr + (size_t)l * DH * DH;
        const float* bl_l = bl + (size_t)l * DH;
        const float* br_l = br + (size_t)l * DH;
        const float* at_l = att + (size_t)l * DH;
        const float* bi_l = bias + (size_t)l * DH;

        // Projections (WMMA)
        gemm_wmma_f16<<<gemm_blocks, 256, 0, stream>>>(xcur, Wl_l, bl_l, xl, DH);
        gemm_wmma_f16<<<gemm_blocks, 256, 0, stream>>>(xcur, Wr_l, br_l, xr, DH);

        // Softmax + aggregate over edges
        layer_init<<<node_grid, 256, 0, stream>>>(acc, maxv, denom);
        edge_logits<<<edge_wave_grid, 256, 0, stream>>>(xl, xr, at_l, src, dst,
                                                        logit, maxv);
        edge_exp<<<edge_thr_grid, 256, 0, stream>>>(logit, dst, maxv, exv, denom);
        edge_scatter<<<edge_wave_grid, 256, 0, stream>>>(xl, src, dst, exv,
                                                         denom, acc);
        bias_relu<<<node_grid, 256, 0, stream>>>(acc, bi_l, xcur);
    }

    // MLP head: h = xcur@W1.T + b1 ; out = h@W2.T + b2  (reuse xl as h)
    gemm_wmma_f16<<<gemm_blocks, 256, 0, stream>>>(xcur, W1, b1, xl, DH);
    gemm_wmma_f16<<<gemm_blocks, 128, 0, stream>>>(xl, W2, b2, out, DOUT);
}
